// ComplexMessagePassing_84121229459549
// MI455X (gfx1250) — compile-verified, matrix operand order checked by reading, not run on hardware
//
#include <hip/hip_runtime.h>
#include <math.h>

// ---------------------------------------------------------------------------
// ComplexMessagePassing on MI455X (gfx1250, wave32, WMMA)
//
// Roofline: edge MLP GEMMs = 7.9e10 FLOPs -> f16 WMMA (matrix pipe, ~tens of
// us). Data movement ~2 GB: rbf stream 205MB (NT, bypass L2), src gathers
// 820MB (L2-cached), atomic segment-sum ~820MB (agg buffers are 51MB -> L2
// resident), node pass ~150MB. => memory/scatter bound at ~90us @ 23.3 TB/s.
// GEMMs ride WMMA; VALU does activations; single-instruction
// global_atomic_add_f32 (device scope) does the segment_sum.
// ---------------------------------------------------------------------------

typedef __attribute__((ext_vector_type(16))) _Float16 v16h;
typedef __attribute__((ext_vector_type(8)))  _Float16 v8h;
typedef __attribute__((ext_vector_type(8)))  float    v8f;
typedef __attribute__((ext_vector_type(4)))  float    v4f;

#define HID      128
#define EDIM     64
#define NNODES   50000
#define NEDGES   800000
#define LN_EPS   1e-5f

static __device__ __forceinline__ v8f wmma_f16(v16h a, v16h b, v8f c) {
  // D = A(16x32 f16) * B(32x16 f16) + C(16x16 f32)
  return __builtin_amdgcn_wmma_f32_16x16x32_f16(
      /*neg_a=*/false, a, /*neg_b=*/false, b,
      /*c_mod=*/(short)0, c, /*reuse_a=*/false, /*reuse_b=*/false);
}

// Hardware fp32 atomic add, no-return form (STOREcnt), device scope so the
// segment-sum is coherent across WGPs. Avoids any CAS-loop lowering.
static __device__ __forceinline__ void atomic_add_f32(float* p, float v) {
  asm volatile("global_atomic_add_f32 %0, %1, off scope:SCOPE_DEV"
               :: "v"(p), "v"(v) : "memory");
}

static __device__ __forceinline__ v16h pack16(v8h lo, v8h hi) {
  v16h r;
#pragma unroll
  for (int i = 0; i < 8; ++i) { r[i] = lo[i]; r[8 + i] = hi[i]; }
  return r;
}

// CDNA5 16-bit A/B operand layout: for lane half hi = lane>>4, the 16 halves
// of the operand are K = {base+hi*8 .. +7} and K = {base+16+hi*8 .. +7}.
// With row-major [row][K] f16 storage those are two contiguous 16B chunks.
static __device__ __forceinline__ v16h load_ab16(const _Float16* p) {
  return pack16(*(const v8h*)p, *(const v8h*)(p + 16));
}

static __device__ __forceinline__ v16h cvt16(v4f f0, v4f f1, v4f f2, v4f f3) {
  v16h a;
#pragma unroll
  for (int i = 0; i < 4; ++i) {
    a[i]      = (_Float16)f0[i];
    a[4 + i]  = (_Float16)f1[i];
    a[8 + i]  = (_Float16)f2[i];
    a[12 + i] = (_Float16)f3[i];
  }
  return a;
}

// A operand from fp32 rows (mag / agg): regular temporal loads (reused via L2)
static __device__ __forceinline__ v16h load_a_f32(const float* p) {
  return cvt16(*(const v4f*)(p),      *(const v4f*)(p + 4),
               *(const v4f*)(p + 16), *(const v4f*)(p + 20));
}

// A operand from fp32 rows, non-temporal (rbf is streamed exactly once and
// is larger than the 192MB L2 -> don't pollute it).
static __device__ __forceinline__ v16h load_a_f32_nt(const float* p) {
  return cvt16(__builtin_nontemporal_load((const v4f*)(p)),
               __builtin_nontemporal_load((const v4f*)(p + 4)),
               __builtin_nontemporal_load((const v4f*)(p + 16)),
               __builtin_nontemporal_load((const v4f*)(p + 20)));
}

// ---------------------------------------------------------------------------
// Prep kernels
// ---------------------------------------------------------------------------

// dst (f16, [128][K] i.e. N-major/transposed) <- src (f32, [K][128])
__global__ void transpose_to_f16(_Float16* __restrict__ dst,
                                 const float* __restrict__ src, int K) {
  int i = blockIdx.x * blockDim.x + threadIdx.x;
  if (i >= K * HID) return;
  int n = i & (HID - 1);
  int k = i >> 7;
  dst[n * K + k] = (_Float16)src[k * HID + n];
}

__global__ void zero4(float4* __restrict__ p, int n4) {
  int i = blockIdx.x * blockDim.x + threadIdx.x;
  if (i < n4) p[i] = make_float4(0.f, 0.f, 0.f, 0.f);
}

// ---------------------------------------------------------------------------
// Edge kernel: 2x (2-layer MLP via WMMA) + gather + atomic scatter-add.
// Block = 256 threads = 8 waves; each wave owns one 16-edge M-tile per
// iteration and all 8 N-tiles (C = 8 x v8f). Weights (96KB f16) staged in
// LDS once per block; per-wave 16x128 f16 LDS tile re-layouts layer-1 output
// (C layout) into layer-2 A operands.
// ---------------------------------------------------------------------------
__global__ void __launch_bounds__(256, 1)
edge_mlp_scatter(const float* __restrict__ rbf,
                 const float* __restrict__ mag,
                 const float* __restrict__ phase,
                 const int*   __restrict__ edge_index,   // [2][E]
                 const _Float16* __restrict__ wtab,      // WmT1|WmT2|WpT1|WpT2
                 const float* __restrict__ bm1, const float* __restrict__ bm2,
                 const float* __restrict__ bp1, const float* __restrict__ bp2,
                 const float* __restrict__ phase_scale,
                 float* __restrict__ agg_mag,
                 float* __restrict__ agg_phase) {
  extern __shared__ _Float16 lds[];
  _Float16* sW = lds;                                   // 49152 halves (96KB)
  _Float16* sH = lds + 49152 + (threadIdx.x >> 5) * (16 * HID); // per-wave 4KB

  { // cooperative weight staging: 49152 halves as b128 chunks
    const v8h* s = (const v8h*)wtab;
    v8h* d = (v8h*)sW;
    for (int i = threadIdx.x; i < 49152 / 8; i += blockDim.x) d[i] = s[i];
  }
  __syncthreads();

  const int lane = threadIdx.x & 31;
  const int lo = lane & 15, hi = lane >> 4;
  const int wave = threadIdx.x >> 5;
  const float ps = phase_scale[0];

  const int nTiles = NEDGES / 16;
  for (int tile = blockIdx.x * 8 + wave; tile < nTiles; tile += gridDim.x * 8) {
    const int m0 = tile * 16;

    // Edge endpoints for the rows this lane's C fragments touch.
    int esrc[8], edst[8];
#pragma unroll
    for (int r = 0; r < 8; ++r) {
      const int e = m0 + r + 8 * hi;
      esrc[r] = edge_index[e];
      edst[r] = edge_index[NEDGES + e];
    }

#pragma unroll
    for (int p = 0; p < 2; ++p) {  // p==0: magnitude MLP, p==1: phase MLP
      const _Float16* W1 = (p == 0) ? sW          : sW + 24576;
      const _Float16* W2 = (p == 0) ? sW + 8192   : sW + 32768;
      const float*    b1 = (p == 0) ? bm1 : bp1;
      const float*    b2 = (p == 0) ? bm2 : bp2;
      float*         agg = (p == 0) ? agg_mag : agg_phase;

      // ---- layer 1: [16 x 64] @ [64 x 128] ----
      v8f c1[8] = {};
#pragma unroll
      for (int ks = 0; ks < 2; ++ks) {
        const int k0 = ks * 32;
        v16h a = load_a_f32_nt(rbf + (size_t)(m0 + lo) * EDIM + k0 + hi * 8);
#pragma unroll
        for (int t = 0; t < 8; ++t) {
          v16h b = load_ab16(W1 + (t * 16 + lo) * EDIM + k0 + hi * 8);
          c1[t] = wmma_f16(a, b, c1[t]);
        }
      }
      // bias + SiLU -> per-wave LDS H tile ([16][128] f16, row-major)
#pragma unroll
      for (int t = 0; t < 8; ++t) {
        const int d = t * 16 + lo;
        const float bv = b1[d];
#pragma unroll
        for (int r = 0; r < 8; ++r) {
          float x = c1[t][r] + bv;
          sH[(r + 8 * hi) * HID + d] = (_Float16)(x / (1.f + __expf(-x)));
        }
      }

      // ---- layer 2: [16 x 128] @ [128 x 128] ----
      v8f c2[8] = {};
#pragma unroll
      for (int ks = 0; ks < 4; ++ks) {
        const int k0 = ks * 32;
        v16h a = load_ab16(sH + lo * HID + k0 + hi * 8);
#pragma unroll
        for (int t = 0; t < 8; ++t) {
          v16h b = load_ab16(W2 + (t * 16 + lo) * HID + k0 + hi * 8);
          c2[t] = wmma_f16(a, b, c2[t]);
        }
      }

      // ---- activation + gather mag/phase[src] + scatter-add to agg[dst] ----
#pragma unroll
      for (int t = 0; t < 8; ++t) {
        const int d = t * 16 + lo;
        const float bv = b2[d];
#pragma unroll
        for (int r = 0; r < 8; ++r) {
          const float x = c2[t][r] + bv;
          float msg;
          if (p == 0) {
            const float w = (x > 20.f) ? x : __logf(1.f + __expf(x)); // softplus
            msg = w * mag[(size_t)esrc[r] * HID + d];
          } else {
            // tanh(x) = 1 - 2/(exp(2x)+1): one v_exp_f32 + one rcp
            const float w = 1.f - 2.f / (__expf(2.f * x) + 1.f);
            msg = w * ps + phase[(size_t)esrc[r] * HID + d];
          }
          atomic_add_f32(agg + (size_t)edst[r] * HID + d, msg); // segment_sum
        }
      }
    }
  }
}

// ---------------------------------------------------------------------------
// Node kernel: gate GEMM [N,256]@[256,128] via WMMA, sigmoid, then
// LayerNorm with wave32 shuffle butterflies (rows of the C tile live in
// 16-lane halves; xor 1/2/4/8 reduces each row), plus phase update.
// ---------------------------------------------------------------------------
__global__ void __launch_bounds__(256, 1)
gate_ln(const float* __restrict__ mag, const float* __restrict__ phase,
        const float* __restrict__ agg_mag, const float* __restrict__ agg_phase,
        const _Float16* __restrict__ WgT,   // [128][256] f16 (N-major)
        const float* __restrict__ bg,
        const float* __restrict__ ln_g, const float* __restrict__ ln_b,
        float* __restrict__ out_mag, float* __restrict__ out_phase) {
  extern __shared__ _Float16 sWg[];         // 32768 halves (64KB)
  {
    const v8h* s = (const v8h*)WgT;
    v8h* d = (v8h*)sWg;
    for (int i = threadIdx.x; i < (HID * 2 * HID) / 8; i += blockDim.x) d[i] = s[i];
  }
  __syncthreads();

  const int lane = threadIdx.x & 31;
  const int lo = lane & 15, hi = lane >> 4;
  const int wave = threadIdx.x >> 5;

  const int nTiles = NNODES / 16;
  for (int tile = blockIdx.x * 8 + wave; tile < nTiles; tile += gridDim.x * 8) {
    const int m0 = tile * 16;

    v8f g[8] = {};
#pragma unroll
    for (int ks = 0; ks < 8; ++ks) {        // K = 256 = concat(mag, agg_mag)
      const int k0 = ks * 32;
      const float* src = (k0 < HID) ? mag : agg_mag;
      const int kk = k0 & (HID - 1);
      v16h a = load_a_f32(src + (size_t)(m0 + lo) * HID + kk + hi * 8);
#pragma unroll
      for (int t = 0; t < 8; ++t) {
        v16h b = load_ab16(sWg + (t * 16 + lo) * (2 * HID) + k0 + hi * 8);
        g[t] = wmma_f16(a, b, g[t]);
      }
    }

    // gate = sigmoid(g + bg); pre = mag + gate*agg_mag; accumulate row stats
    v8f pre[8], gt[8];
    float s1[8], s2[8];
#pragma unroll
    for (int r = 0; r < 8; ++r) { s1[r] = 0.f; s2[r] = 0.f; }
#pragma unroll
    for (int t = 0; t < 8; ++t) {
      const int d = t * 16 + lo;
      const float bv = bg[d];
#pragma unroll
      for (int r = 0; r < 8; ++r) {
        const size_t node = (size_t)(m0 + r + 8 * hi);
        const float gate = 1.f / (1.f + __expf(-(g[t][r] + bv)));
        const float pr = mag[node * HID + d] + gate * agg_mag[node * HID + d];
        gt[t][r] = gate;
        pre[t][r] = pr;
        s1[r] += pr;
        s2[r] += pr * pr;
      }
    }
    // reduce across the 16 lanes of each half-wave (one row per (r,hi))
#pragma unroll
    for (int r = 0; r < 8; ++r) {
#pragma unroll
      for (int m = 1; m <= 8; m <<= 1) {
        s1[r] += __shfl_xor(s1[r], m, 32);
        s2[r] += __shfl_xor(s2[r], m, 32);
      }
    }
#pragma unroll
    for (int t = 0; t < 8; ++t) {
      const int d = t * 16 + lo;
      const float ga = ln_g[d], be = ln_b[d];
#pragma unroll
      for (int r = 0; r < 8; ++r) {
        const size_t node = (size_t)(m0 + r + 8 * hi);
        const float mu  = s1[r] * (1.f / HID);
        const float var = s2[r] * (1.f / HID) - mu * mu;
        const float rs  = rsqrtf(var + LN_EPS);
        // outputs are write-once, never re-read: non-temporal stores
        __builtin_nontemporal_store((pre[t][r] - mu) * rs * ga + be,
                                    &out_mag[node * HID + d]);
        __builtin_nontemporal_store(phase[node * HID + d] +
                                        gt[t][r] * agg_phase[node * HID + d],
                                    &out_phase[node * HID + d]);
      }
    }
  }
}

// ---------------------------------------------------------------------------
// Host launcher
// ---------------------------------------------------------------------------
extern "C" void kernel_launch(void* const* d_in, const int* in_sizes, int n_in,
                              void* d_out, int out_size, void* d_ws, size_t ws_size,
                              hipStream_t stream) {
  (void)in_sizes; (void)n_in; (void)out_size; (void)ws_size;
  const float* mag   = (const float*)d_in[0];
  const float* phase = (const float*)d_in[1];
  const float* rbf   = (const float*)d_in[2];
  const float* Wm1   = (const float*)d_in[3];
  const float* bm1   = (const float*)d_in[4];
  const float* Wm2   = (const float*)d_in[5];
  const float* bm2   = (const float*)d_in[6];
  const float* Wp1   = (const float*)d_in[7];
  const float* bp1   = (const float*)d_in[8];
  const float* Wp2   = (const float*)d_in[9];
  const float* bp2   = (const float*)d_in[10];
  const float* Wg    = (const float*)d_in[11];
  const float* bg    = (const float*)d_in[12];
  const float* ln_g  = (const float*)d_in[13];
  const float* ln_b  = (const float*)d_in[14];
  const float* pscal = (const float*)d_in[15];
  const int*   eidx  = (const int*)d_in[16];

  // Workspace layout: f16 transposed weights (160KB, padded to 256KB),
  // then agg_mag / agg_phase (25.6MB each, contiguous for one zero pass).
  char* ws = (char*)d_ws;
  _Float16* WmT1 = (_Float16*)ws;            // 64*128
  _Float16* WmT2 = WmT1 + 64 * 128;          // 128*128
  _Float16* WpT1 = WmT2 + 128 * 128;         // 64*128
  _Float16* WpT2 = WpT1 + 64 * 128;          // 128*128
  _Float16* WgT  = WpT2 + 128 * 128;         // 128*256
  float* agg_mag   = (float*)(ws + 256 * 1024);
  float* agg_phase = agg_mag + (size_t)NNODES * HID;

  // 1) convert + transpose weights to f16 [N][K]
  transpose_to_f16<<<(64 * 128 + 255) / 256, 256, 0, stream>>>(WmT1, Wm1, 64);
  transpose_to_f16<<<(128 * 128 + 255) / 256, 256, 0, stream>>>(WmT2, Wm2, 128);
  transpose_to_f16<<<(64 * 128 + 255) / 256, 256, 0, stream>>>(WpT1, Wp1, 64);
  transpose_to_f16<<<(128 * 128 + 255) / 256, 256, 0, stream>>>(WpT2, Wp2, 128);
  transpose_to_f16<<<(256 * 128 + 255) / 256, 256, 0, stream>>>(WgT, Wg, 256);

  // 2) zero the two aggregation buffers (atomics accumulate into them)
  const int n4 = NNODES * HID * 2 / 4;
  zero4<<<(n4 + 255) / 256, 256, 0, stream>>>((float4*)agg_mag, n4);

  // 3) edge MLPs + segment_sum (atomic scatter)
  const size_t edge_lds = (size_t)(49152 + 8 * 16 * HID) * sizeof(_Float16); // 128KB
  edge_mlp_scatter<<<512, 256, edge_lds, stream>>>(
      rbf, mag, phase, eidx, WmT1, bm1, bm2, bp1, bp2, pscal,
      agg_mag, agg_phase);

  // 4) gate GEMM + LayerNorm + phase update
  float* out_mag   = (float*)d_out;
  float* out_phase = out_mag + (size_t)NNODES * HID;
  const size_t gate_lds = (size_t)(HID * 2 * HID) * sizeof(_Float16);        // 64KB
  gate_ln<<<392, 256, gate_lds, stream>>>(
      mag, phase, agg_mag, agg_phase, WgT, bg, ln_g, ln_b, out_mag, out_phase);
}
